// Decoder_17248588660774
// MI455X (gfx1250) — matrix operand order repaired and checked by reading, not compiled, not run on hardware
//
#include <hip/hip_runtime.h>

typedef __bf16 bf16_t;
typedef __attribute__((ext_vector_type(16))) __bf16 v16bf;
typedef __attribute__((ext_vector_type(8)))  float  v8f;
typedef __attribute__((ext_vector_type(4)))  unsigned int v4u;
typedef __attribute__((ext_vector_type(8)))  int v8i;
typedef __attribute__((ext_vector_type(4)))  int v4i;

static __device__ inline bf16_t f2bf(float f) {
  unsigned u = __builtin_bit_cast(unsigned, f);
  unsigned r = u + 0x7FFFu + ((u >> 16) & 1u);
  unsigned short h = (unsigned short)(r >> 16);
  return __builtin_bit_cast(bf16_t, h);
}

static __device__ inline float fast_tanh(float x) {
  float e = __expf(2.f * x);
  return (e - 1.f) / (e + 1.f);
}
static __device__ inline float fast_sig(float x) {
  return 1.f / (1.f + __expf(-x));
}

// ---------------- WMMA fragment helpers (ISA 7.12.2 layouts, wave32) -------

// A fragment 16x32 bf16 from row-major [rows x ldk]:
// lanes 0-15: rows M=0..15, K = {0..7, 16..23}; lanes 16-31: same rows, K = {8..15, 24..31}
static __device__ inline v16bf load_a_frag(const bf16_t* A, int row0, int k0, int ldk) {
  int lane = threadIdx.x & 31;
  const bf16_t* p = A + (row0 + (lane & 15)) * ldk + k0 + ((lane >> 4) & 1) * 8;
  v16bf a;
  ((uint4*)&a)[0] = *(const uint4*)p;         // K block +0..7
  ((uint4*)&a)[1] = *(const uint4*)(p + 16);  // K block +16..23
  return a;
}

// B fragment 32x16 bf16 where source is stored [N x K] row-major (i.e. W for X*W^T,
// or an LDS tile stored transposed as [n][k]):
// lanes 0-15: N=lane, K = {0..7, 16..23} ; lanes 16-31: N=lane-16, K = {8..15, 24..31}
static __device__ inline v16bf load_b_nk(const bf16_t* W, int n0, int k0, int ldk) {
  int lane = threadIdx.x & 31;
  const bf16_t* p = W + (size_t)(n0 + (lane & 15)) * ldk + k0 + ((lane >> 4) & 1) * 8;
  v16bf b;
  ((uint4*)&b)[0] = *(const uint4*)p;
  ((uint4*)&b)[1] = *(const uint4*)(p + 16);
  return b;
}

static __device__ inline v8f wmma_bf16(v16bf a, v16bf b, v8f c) {
  return __builtin_amdgcn_wmma_f32_16x16x32_bf16(false, a, false, b, (short)0, c,
                                                 false, false);
}

// ---------------- TDM: async tensor load of a 2D fp32 tile into LDS --------
// Tile: cols x rows, element stride 1 along cols, row stride = tensor_w elements.
// D# packing per CDNA5 ISA 8.3/8.4 (group0: count|lds_addr|global_addr|type=2,
// group1: data_size=4B, tensor_dim0/1, tile_dim0/1, tensor_dim0_stride).
// This toolchain exposes the 6-arg builtin: (g0, g1, g2, g3, g_extra, cpol).
static __device__ inline void tdm_load_tile_f32(const float* gsrc, unsigned lds_off,
                                                unsigned tensor_w, unsigned rows,
                                                unsigned cols) {
  unsigned long long ga = (unsigned long long)(size_t)gsrc;
  v4u g0;
  g0.x = 1u;                                                   // count=1 (valid)
  g0.y = lds_off;                                              // lds_addr
  g0.z = (unsigned)ga;                                         // global_addr[31:0]
  g0.w = (unsigned)((ga >> 32) & 0x01FFFFFFu) | (2u << 30);    // addr[56:32] | type=2
  unsigned long long st0 = (unsigned long long)tensor_w;       // dim0 stride (elems)
  unsigned long long q0 = (2ull << 16)                         // data_size = 4 bytes
                        | ((unsigned long long)(tensor_w & 0xFFFFu) << 48);
  unsigned long long q1 = (unsigned long long)(tensor_w >> 16)       // tensor_dim0 hi
                        | ((unsigned long long)rows << 16)           // tensor_dim1
                        | ((unsigned long long)cols << 48);          // tile_dim0
  unsigned long long q2 = (unsigned long long)rows                   // tile_dim1
                        | ((st0 & 0xFFFFFFFFull) << 32);             // stride0[31:0]
  unsigned long long q3 = (st0 >> 32) & 0xFFFFull;                   // stride0[47:32]
  v8i g1;
  g1[0] = (int)(unsigned)q0; g1[1] = (int)(q0 >> 32);
  g1[2] = (int)(unsigned)q1; g1[3] = (int)(q1 >> 32);
  g1[4] = (int)(unsigned)q2; g1[5] = (int)(q2 >> 32);
  g1[6] = (int)(unsigned)q3; g1[7] = (int)(q3 >> 32);
  v4i z4 = {0, 0, 0, 0};
  v8i z8 = {0, 0, 0, 0, 0, 0, 0, 0};
  __builtin_amdgcn_tensor_load_to_lds(g0, g1, z4, z4, z8, 0);
}

// ---------------- generic GEMM: C[MxN] = X[MxK](bf16) * W[NxK]^T -----------
__global__ void k_gemm(const bf16_t* __restrict__ X, const bf16_t* __restrict__ W,
                       float* __restrict__ C, int N, int K) {
  int n0 = blockIdx.x * 16, m0 = blockIdx.y * 16;
  v8f acc = {};
  for (int k0 = 0; k0 < K; k0 += 32) {
    v16bf a = load_a_frag(X, m0, k0, K);
    v16bf b = load_b_nk(W, n0, k0, K);
    acc = wmma_bf16(a, b, acc);
  }
  int lane = threadIdx.x & 31;
  int n = n0 + (lane & 15);
  int mo = m0 + ((lane >> 4) ? 8 : 0);
#pragma unroll
  for (int r = 0; r < 8; ++r) C[(size_t)(mo + r) * N + n] = acc[r];
}

// ---------------- small prep kernels ---------------------------------------
__global__ void k_f2bf(const float* __restrict__ s, bf16_t* __restrict__ d, int n) {
  int i = blockIdx.x * 256 + threadIdx.x;
  if (i < n) d[i] = f2bf(s[i]);
}

// G[p,k] = sum_c U_f[p,c] * convw[c, k]  (k < taps), zero-padded to KPAD; bf16 out
__global__ void k_convfold(const float* __restrict__ Uf, const float* __restrict__ cw,
                           bf16_t* __restrict__ G, int taps, int KPAD) {
  int idx = blockIdx.x * 256 + threadIdx.x;
  if (idx >= 512 * KPAD) return;
  int p = idx / KPAD, k = idx - p * KPAD;
  float acc = 0.f;
  if (k < taps)
    for (int c = 0; c < 256; ++c) acc += Uf[p * 256 + c] * cw[c * taps + k];
  G[idx] = f2bf(acc);
}

// fb[p] = sum_c U_f[p,c] * conv_b[c]
__global__ void k_fbias(const float* __restrict__ Uf, const float* __restrict__ cb,
                        float* __restrict__ fb) {
  int p = blockIdx.x * 256 + threadIdx.x;
  if (p >= 512) return;
  float a = 0.f;
  for (int c = 0; c < 256; ++c) a += Uf[p * 256 + c] * cb[c];
  fb[p] = a;
}

__global__ void k_gather(const int* __restrict__ x, const float* __restrict__ tab,
                         const float* __restrict__ hidden, float* __restrict__ emb_f,
                         bf16_t* __restrict__ emb_b, bf16_t* __restrict__ hid_b) {
  int b = blockIdx.x, e = threadIdx.x;
  int cls = x[b];
  float v = tab[cls * 256 + e];
  emb_f[b * 256 + e] = v;
  emb_b[b * 256 + e] = f2bf(v);
  hid_b[b * 256 + e] = f2bf(hidden[b * 256 + e]);
}

// GRU gates from precomputed gi/gh (64 x 768 each)
__global__ void k_gates(const float* __restrict__ gi, const float* __restrict__ gh,
                        const float* __restrict__ bih, const float* __restrict__ bhh,
                        const float* __restrict__ hprev, float* __restrict__ out_f,
                        bf16_t* __restrict__ out_b) {
  int b = blockIdx.x, h = threadIdx.x;
  size_t o = (size_t)b * 768;
  float r = fast_sig(gi[o + h] + bih[h] + gh[o + h] + bhh[h]);
  float z = fast_sig(gi[o + 256 + h] + bih[256 + h] + gh[o + 256 + h] + bhh[256 + h]);
  float n = fast_tanh(gi[o + 512 + h] + bih[512 + h] +
                      r * (gh[o + 512 + h] + bhh[512 + h]));
  float v = (1.f - z) * n + z * hprev[(size_t)b * 256 + h];
  out_f[(size_t)b * 256 + h] = v;
  out_b[(size_t)b * 256 + h] = f2bf(v);
}

// ---------------- fused attention energy -----------------------------------
// e[b,l] = sum_p nu[p] * tanh( (U_a x)[p,l] + u_pred[b,p] + fb[p] + (G patch)[p,l] )
// x tiles are streamed Global->LDS by the Tensor Data Mover, double-buffered on
// TENSORcnt, while the 8 waves run the bf16 WMMA stack.
template <int H, int W, int KS, int KPAD>
__global__ void k_attn_e(const float* __restrict__ xres, const float* __restrict__ asum,
                         const bf16_t* __restrict__ Ua, const bf16_t* __restrict__ G,
                         const float* __restrict__ nu, const float* __restrict__ upred,
                         const float* __restrict__ fb, float* __restrict__ e_g) {
  constexpr int L = H * W;
  constexpr int PAD = (KS - 1) / 2;
  constexpr int TAPS = KS * KS;
  const int b = blockIdx.x;
  const int tid = threadIdx.x;
  const int lane = tid & 31;
  const int wave = tid >> 5;

  __shared__ __align__(16) float stage[2][256 * 16];  // raw fp32 tiles from TDM
  __shared__ __align__(16) bf16_t xs[16 * 256];       // [n][k]  k = channel
  __shared__ __align__(16) bf16_t ps[16 * KPAD];      // [n][k]  k = conv tap
  __shared__ float ups[512];
  __shared__ float nus[512];
  __shared__ float es[16];

  for (int p = tid; p < 512; p += 256) {
    ups[p] = upred[(size_t)b * 512 + p] + fb[p];
    nus[p] = nu[p];
  }

  const int tilesTotal = L / 16;
  const int tpc = tilesTotal / gridDim.y;
  const int t0 = blockIdx.y * tpc;
  const int tEnd = t0 + tpc;
  const float* xbase = xres + (size_t)b * 256 * L;

  if (wave == 0)
    tdm_load_tile_f32(xbase + t0 * 16, (unsigned)(size_t)&stage[t0 & 1][0], L, 256, 16);

  for (int t = t0; t < tEnd; ++t) {
    const int l0 = t * 16;
    if (wave == 0) {
      if (t + 1 < tEnd) {
        tdm_load_tile_f32(xbase + (t + 1) * 16,
                          (unsigned)(size_t)&stage[(t + 1) & 1][0], L, 256, 16);
        __builtin_amdgcn_s_wait_tensorcnt(1);  // tile t landed (TDM is in-order)
      } else {
        __builtin_amdgcn_s_wait_tensorcnt(0);
      }
    }
    if (tid < 16) es[tid] = 0.f;

    // build im2col patch tile ps[n][k] from alpha_sum (VALU path, tiny)
    for (int idx = tid; idx < KPAD * 16; idx += 256) {
      int k = idx & (KPAD - 1);
      int n = idx / KPAD;
      float v = 0.f;
      if (k < TAPS) {
        int l = l0 + n;
        int y = l / W, x = l - y * W;
        int yy = y + (k / KS) - PAD;
        int xx = x + (k % KS) - PAD;
        if (yy >= 0 && yy < H && xx >= 0 && xx < W)
          v = asum[(size_t)b * L + yy * W + xx];
      }
      ps[n * KPAD + k] = f2bf(v);
    }
    __syncthreads();  // publish DMA'd stage[t&1] + ps to all waves

    // convert fp32 stage tile [k][n] -> bf16 xs [n][k]
    {
      const float* sp = &stage[t & 1][tid * 16];
#pragma unroll
      for (int n = 0; n < 16; ++n) xs[n * 256 + tid] = f2bf(sp[n]);
    }
    __syncthreads();  // xs ready

    // each wave: 4 p-tiles of 16 rows; K=256 over U_a + K=KPAD over G
    for (int pt = wave; pt < 32; pt += 8) {
      const int p0 = pt * 16;
      v8f acc = {};
#pragma unroll
      for (int k0 = 0; k0 < 256; k0 += 32)
        acc = wmma_bf16(load_a_frag(Ua, p0, k0, 256), load_b_nk(xs, 0, k0, 256), acc);
#pragma unroll
      for (int k0 = 0; k0 < KPAD; k0 += 32)
        acc = wmma_bf16(load_a_frag(G, p0, k0, KPAD), load_b_nk(ps, 0, k0, KPAD), acc);

      const int mo = p0 + ((lane >> 4) ? 8 : 0);
      float pe = 0.f;
#pragma unroll
      for (int r = 0; r < 8; ++r)
        pe += nus[mo + r] * fast_tanh(acc[r] + ups[mo + r]);
      pe += __shfl_xor(pe, 16);
      if (lane < 16) atomicAdd(&es[lane], pe);
    }
    __syncthreads();
    if (tid < 16) e_g[(size_t)b * L + l0 + tid] = es[tid];
  }
}

// ---------------- softmax + context ----------------------------------------
__global__ void k_attn_sm(const float* __restrict__ e_g, const float* __restrict__ xres,
                          float* __restrict__ ctx_f, bf16_t* __restrict__ ctx_b, int L,
                          int ctx_off) {
  const int b = blockIdx.x, tid = threadIdx.x, lane = tid & 31, wave = tid >> 5;
  __shared__ float sal[1024];
  __shared__ float red[8];

  const float* xr = xres + ((size_t)b * 256 + tid) * L;
  __builtin_prefetch(xr, 0, 1);  // -> global_prefetch_b8

  float m = -1e30f;
  for (int l = tid; l < L; l += 256) {
    float v = e_g[(size_t)b * L + l];
    sal[l] = v;
    m = fmaxf(m, v);
  }
#pragma unroll
  for (int off = 16; off; off >>= 1) m = fmaxf(m, __shfl_xor(m, off));
  if (lane == 0) red[wave] = m;
  __syncthreads();
  float mm = red[0];
#pragma unroll
  for (int i = 1; i < 8; ++i) mm = fmaxf(mm, red[i]);
  __syncthreads();

  float s = 0.f;
  for (int l = tid; l < L; l += 256) {
    float v = __expf(sal[l] - mm);
    sal[l] = v;
    s += v;
  }
#pragma unroll
  for (int off = 16; off; off >>= 1) s += __shfl_xor(s, off);
  if (lane == 0) red[wave] = s;
  __syncthreads();
  float st = 0.f;
#pragma unroll
  for (int i = 0; i < 8; ++i) st += red[i];
  float inv = 1.f / st;

  float acc = 0.f;
  for (int l = 0; l < L; ++l) acc += sal[l] * xr[l];
  acc *= inv;
  ctx_f[(size_t)b * 512 + ctx_off + tid] = acc;
  ctx_b[(size_t)b * 512 + ctx_off + tid] = f2bf(acc);
}

// ---------------- output head: pre = emb + o1 + o2, pairwise max -----------
__global__ void k_headmax(const float* __restrict__ emb, const float* __restrict__ o1,
                          const float* __restrict__ o2, bf16_t* __restrict__ mp) {
  int idx = blockIdx.x * 256 + threadIdx.x;  // b*128 + j
  if (idx >= 64 * 128) return;
  int b = idx >> 7, j = idx & 127;
  size_t o = (size_t)b * 256 + 2 * j;
  float a0 = emb[o] + o1[o] + o2[o];
  float a1 = emb[o + 1] + o1[o + 1] + o2[o + 1];
  mp[idx] = f2bf(fmaxf(a0, a1));
}

// ===========================================================================
extern "C" void kernel_launch(void* const* d_in, const int* in_sizes, int n_in,
                              void* d_out, int out_size, void* d_ws, size_t ws_size,
                              hipStream_t stream) {
  (void)in_sizes; (void)n_in; (void)out_size; (void)ws_size;
  const int* x = (const int*)d_in[0];
  const float* hidden = (const float*)d_in[1];
  const float* low_res = (const float*)d_in[2];
  const float* high_res = (const float*)d_in[3];
  const float* asum_low = (const float*)d_in[4];
  const float* asum_high = (const float*)d_in[5];
  const float* embedding = (const float*)d_in[6];
  const float* g1Wih = (const float*)d_in[7];
  const float* g1Whh = (const float*)d_in[8];
  const float* g1bih = (const float*)d_in[9];
  const float* g1bhh = (const float*)d_in[10];
  const float* g2Wih = (const float*)d_in[11];
  const float* g2Whh = (const float*)d_in[12];
  const float* g2bih = (const float*)d_in[13];
  const float* g2bhh = (const float*)d_in[14];
  const float* clw = (const float*)d_in[15];
  const float* clb = (const float*)d_in[16];
  const float* chw = (const float*)d_in[17];
  const float* chb = (const float*)d_in[18];
  const float* UaL = (const float*)d_in[19];
  const float* UfL = (const float*)d_in[20];
  const float* nuL = (const float*)d_in[21];
  const float* UaH = (const float*)d_in[22];
  const float* UfH = (const float*)d_in[23];
  const float* nuH = (const float*)d_in[24];
  const float* Upred = (const float*)d_in[25];
  const float* Ws = (const float*)d_in[26];
  const float* Wc = (const float*)d_in[27];
  const float* Wo = (const float*)d_in[28];

  float* out = (float*)d_out;        // 64 x 144
  float* newh_out = out + 64 * 144;  // 64 x 256

  char* w = (char*)d_ws;
  auto alloc = [&](size_t bytes) -> char* {
    char* p = w;
    w += (bytes + 255) & ~(size_t)255;
    return p;
  };
  bf16_t* UaL_b = (bf16_t*)alloc(512 * 256 * 2);
  bf16_t* UaH_b = (bf16_t*)alloc(512 * 256 * 2);
  bf16_t* Up_b  = (bf16_t*)alloc(512 * 256 * 2);
  bf16_t* W1i_b = (bf16_t*)alloc(768 * 256 * 2);
  bf16_t* W1h_b = (bf16_t*)alloc(768 * 256 * 2);
  bf16_t* W2i_b = (bf16_t*)alloc(768 * 512 * 2);
  bf16_t* W2h_b = (bf16_t*)alloc(768 * 256 * 2);
  bf16_t* Ws_b  = (bf16_t*)alloc(256 * 256 * 2);
  bf16_t* Wc_b  = (bf16_t*)alloc(256 * 512 * 2);
  bf16_t* Wo_b  = (bf16_t*)alloc(144 * 128 * 2);
  bf16_t* GL_b  = (bf16_t*)alloc(512 * 128 * 2);
  bf16_t* GH_b  = (bf16_t*)alloc(512 * 64 * 2);
  float* fbL    = (float*)alloc(512 * 4);
  float* fbH    = (float*)alloc(512 * 4);
  float* emb_f  = (float*)alloc(64 * 256 * 4);
  bf16_t* emb_b = (bf16_t*)alloc(64 * 256 * 2);
  bf16_t* hid_b = (bf16_t*)alloc(64 * 256 * 2);
  float* gi1    = (float*)alloc(64 * 768 * 4);
  float* gh1    = (float*)alloc(64 * 768 * 4);
  float* pred_f = (float*)alloc(64 * 256 * 4);
  bf16_t* pred_b= (bf16_t*)alloc(64 * 256 * 2);
  float* upred  = (float*)alloc(64 * 512 * 4);
  float* eL     = (float*)alloc(64 * 256 * 4);
  float* eH     = (float*)alloc(64 * 1024 * 4);
  float* ctx_f  = (float*)alloc(64 * 512 * 4);
  bf16_t* ctx_b = (bf16_t*)alloc(64 * 512 * 2);
  float* gi2    = (float*)alloc(64 * 768 * 4);
  float* gh2    = (float*)alloc(64 * 768 * 4);
  bf16_t* newh_b= (bf16_t*)alloc(64 * 256 * 2);
  float* o1     = (float*)alloc(64 * 256 * 4);
  float* o2     = (float*)alloc(64 * 256 * 4);
  bf16_t* mp_b  = (bf16_t*)alloc(64 * 128 * 2);

  auto cvt = [&](const float* s, bf16_t* d, int n) {
    k_f2bf<<<(n + 255) / 256, 256, 0, stream>>>(s, d, n);
  };
  // weight conversions (one-time cost; everything then sits in the 192MB L2)
  cvt(UaL, UaL_b, 512 * 256);
  cvt(UaH, UaH_b, 512 * 256);
  cvt(Upred, Up_b, 512 * 256);
  cvt(g1Wih, W1i_b, 768 * 256);
  cvt(g1Whh, W1h_b, 768 * 256);
  cvt(g2Wih, W2i_b, 768 * 512);
  cvt(g2Whh, W2h_b, 768 * 256);
  cvt(Ws, Ws_b, 256 * 256);
  cvt(Wc, Wc_b, 256 * 512);
  cvt(Wo, Wo_b, 144 * 128);
  // fold coverage conv into u_f GEMM:  G = U_f @ Wconv,  fb = U_f @ conv_b
  k_convfold<<<(512 * 128 + 255) / 256, 256, 0, stream>>>(UfL, clw, GL_b, 121, 128);
  k_convfold<<<(512 * 64 + 255) / 256, 256, 0, stream>>>(UfH, chw, GH_b, 49, 64);
  k_fbias<<<2, 256, 0, stream>>>(UfL, clb, fbL);
  k_fbias<<<2, 256, 0, stream>>>(UfH, chb, fbH);

  // embedding gather + bf16 copies
  k_gather<<<64, 256, 0, stream>>>(x, embedding, hidden, emb_f, emb_b, hid_b);

  // GRU1: gi = emb @ W_ih^T, gh = h @ W_hh^T, then gates
  k_gemm<<<dim3(48, 4), 32, 0, stream>>>(emb_b, W1i_b, gi1, 768, 256);
  k_gemm<<<dim3(48, 4), 32, 0, stream>>>(hid_b, W1h_b, gh1, 768, 256);
  k_gates<<<64, 256, 0, stream>>>(gi1, gh1, g1bih, g1bhh, hidden, pred_f, pred_b);

  // u_pred = pred @ U_pred^T
  k_gemm<<<dim3(32, 4), 32, 0, stream>>>(pred_b, Up_b, upred, 512, 256);

  // attention energies (bulk of the FLOPs, bf16 WMMA + TDM-streamed tiles)
  k_attn_e<16, 16, 11, 128><<<dim3(64, 2), 256, 0, stream>>>(
      low_res, asum_low, UaL_b, GL_b, nuL, upred, fbL, eL);
  k_attn_e<32, 32, 7, 64><<<dim3(64, 8), 256, 0, stream>>>(
      high_res, asum_high, UaH_b, GH_b, nuH, upred, fbH, eH);

  // softmax + context vectors
  k_attn_sm<<<64, 256, 0, stream>>>(eL, low_res, ctx_f, ctx_b, 256, 0);
  k_attn_sm<<<64, 256, 0, stream>>>(eH, high_res, ctx_f, ctx_b, 1024, 256);

  // GRU2
  k_gemm<<<dim3(48, 4), 32, 0, stream>>>(ctx_b, W2i_b, gi2, 768, 512);
  k_gemm<<<dim3(48, 4), 32, 0, stream>>>(pred_b, W2h_b, gh2, 768, 256);
  k_gates<<<64, 256, 0, stream>>>(gi2, gh2, g2bih, g2bhh, pred_f, newh_out, newh_b);

  // output head
  k_gemm<<<dim3(16, 4), 32, 0, stream>>>(newh_b, Ws_b, o1, 256, 256);
  k_gemm<<<dim3(16, 4), 32, 0, stream>>>(ctx_b, Wc_b, o2, 256, 512);
  k_headmax<<<32, 256, 0, stream>>>(emb_f, o1, o2, mp_b);
  k_gemm<<<dim3(9, 4), 32, 0, stream>>>(mp_b, Wo_b, out, 144, 128);
}
